// SSD_403726926104
// MI455X (gfx1250) — compile-verified, matrix operand order checked by reading, not run on hardware
//
#include <hip/hip_runtime.h>

// ---------------------------------------------------------------------------
// SSD head for MI455X (gfx1250, wave32).
// Convs as implicit GEMM on V_WMMA_F32_16X16X32_BF16:
//   M = 16 output pixels, N = 16 output channels, K = C*9.
// Weights pre-packed in B-fragment lane order; per workgroup they are staged
// into LDS with double-buffered GLOBAL_LOAD_ASYNC_TO_LDS_B128 (ASYNCcnt
// pipelined against the WMMA loop), shared by all 4 waves. Activation im2col
// uses clamped unconditional gathers (no EXEC save/restore per tap).
// B-fragment LDS reads are software-pipelined one WMMA ahead.
// ---------------------------------------------------------------------------

#if defined(__HIP_DEVICE_COMPILE__) && \
    __has_builtin(__builtin_amdgcn_global_load_async_to_lds_b128) && \
    __has_builtin(__builtin_amdgcn_s_wait_asynccnt)
#define SSD_HAVE_ASYNC_LDS 1
#else
#define SSD_HAVE_ASYNC_LDS 0
#endif

typedef __bf16 v16bf __attribute__((ext_vector_type(16)));
typedef float  v8f   __attribute__((ext_vector_type(8)));
typedef int    v4i   __attribute__((ext_vector_type(4)));

__device__ __forceinline__ __bf16 f2bf(float f) {
    unsigned u = __builtin_bit_cast(unsigned, f);
    unsigned short h = (unsigned short)((u + 0x7FFFu + ((u >> 16) & 1u)) >> 16);
    return __builtin_bit_cast(__bf16, h);
}

// ---- weight pre-pack: f32 [OC][C][3][3] -> bf16 bits in B-fragment order ---
// layout: [(kb*nOct + oct)*512 + lane*16 + j]
// fragment element (lane, j):  K = kb*32 + (j/8)*16 + (lane/16)*8 + (j%8)
//                              N(oc) = oct*16 + (lane&15)
__global__ void __launch_bounds__(256)
pack_weights_k(const float* __restrict__ w, unsigned short* __restrict__ wpk,
               int C, int OC, int nOct, int total) {
    int idx = blockIdx.x * 256 + threadIdx.x;
    if (idx >= total) return;
    int j    = idx & 15;
    int lane = (idx >> 4) & 31;
    int blk  = idx >> 9;            // kb*nOct + oct
    int oct  = blk % nOct;
    int kb   = blk / nOct;
    int kk   = (kb << 5) + ((j >> 3) << 4) + ((lane >> 4) << 3) + (j & 7);
    int oc   = oct * 16 + (lane & 15);
    float v = 0.0f;
    if (oc < OC) {
        unsigned c   = (unsigned)kk / 9u;
        unsigned rem = (unsigned)kk - c * 9u;
        unsigned r   = rem / 3u;
        unsigned s   = rem - r * 3u;
        v = w[(((size_t)oc * C + c) * 3 + r) * 3 + s];
    }
    unsigned u = __builtin_bit_cast(unsigned, v);
    wpk[idx] = (unsigned short)((u + 0x7FFFu + ((u >> 16) & 1u)) >> 16);
}

// ---- anchor builder (SSD300: 8732 anchors) --------------------------------
__global__ void __launch_bounds__(256)
build_anchors_k(float* __restrict__ anc) {
    int a = blockIdx.x * 256 + threadIdx.x;
    if (a >= 8732) return;
    const int   base[7] = {0, 5776, 7942, 8542, 8692, 8728, 8732};
    const int   fs[6]   = {38, 19, 10, 5, 3, 1};
    const int   kk[6]   = {4, 6, 6, 6, 4, 4};
    const float s0[6]   = {30.f, 60.f, 111.f, 162.f, 213.f, 264.f};
    const float s1[6]   = {60.f, 111.f, 162.f, 213.f, 264.f, 315.f};
    const float st[6]   = {8.f, 16.f, 32.f, 64.f, 100.f, 300.f};
    int L = 0;
    while (a >= base[L + 1]) L++;
    int idx = a - base[L];
    int k   = idx % kk[L];
    int pix = idx / kk[L];
    int x = pix % fs[L], y = pix / fs[L];
    float w, h;
    if (k == 0)      { w = h = s0[L]; }
    else if (k == 1) { w = h = sqrtf(s0[L] * s1[L]); }
    else {
        float r  = (k == 2) ? 2.0f : (k == 3) ? 0.5f : (k == 4) ? 3.0f : (1.0f / 3.0f);
        float sr = sqrtf(r);
        w = s0[L] * sr;
        h = s0[L] / sr;
    }
    anc[a * 4 + 0] = ((float)x + 0.5f) * st[L];
    anc[a * 4 + 1] = ((float)y + 0.5f) * st[L];
    anc[a * 4 + 2] = w;
    anc[a * 4 + 3] = h;
}

// ---- implicit-GEMM 3x3 SAME conv via WMMA ---------------------------------
// Block = (32,4): 4 waves share double-buffered weight tiles in LDS.
template <int NOCT>
__global__ void __launch_bounds__(128)
conv3_wmma(const float* __restrict__ in, const unsigned short* __restrict__ wpk,
           const float* __restrict__ bias, float* __restrict__ out,
           int C, int H, int W, int OC, int Kanch, int ncols, int anchor_base) {
    constexpr int CHUNK = 4;                   // K-blocks staged per round
    const int Bt = 32, A = 8732;
    const int lane = threadIdx.x;
    const int wid  = threadIdx.y;
    const int tid  = wid * 32 + lane;
    const int HW = H * W;
    const int P = Bt * HW;
    const int p0 = (blockIdx.x * 4 + wid) * 16;   // no early return: barriers below

    __shared__ __align__(32) unsigned short sbuf[2][CHUNK * NOCT * 512];

    // A-fragment lane coordinates (16-bit A 16x32 layout, ISA 7.12.2)
    const int mrow = lane & 15;                // M (pixel within tile)
    const int hi8  = (lane >> 4) << 3;         // K sub-offset for upper lanes
    int p = p0 + mrow;
    bool pvalid = p < P;
    int pb = 0, py = 0, px = 0;
    if (pvalid) { pb = p / HW; int r = p - pb * HW; py = r / W; px = r - py * W; }
    const float* inb = in + (size_t)pb * C * HW;

    const int nkb = (C * 9) >> 5;              // K-blocks of 32 (multiple of 4 here)
    const int nch = nkb / CHUNK;

    v8f acc[NOCT];
#pragma unroll
    for (int o = 0; o < NOCT; ++o) acc[o] = v8f{};

    auto stage = [&](int ci, int which) {
        v4i* gsrc = (v4i*)(wpk + (size_t)ci * (CHUNK * NOCT * 512));
        v4i* ldst = (v4i*)(&sbuf[which][0]);
#if SSD_HAVE_ASYNC_LDS
#pragma unroll
        for (int t = tid; t < CHUNK * NOCT * 64; t += 128) {
            __builtin_amdgcn_global_load_async_to_lds_b128(
                (__attribute__((address_space(1))) v4i*)(gsrc + t),
                (__attribute__((address_space(3))) v4i*)(ldst + t), 0, 0);
        }
#else
#pragma unroll
        for (int t = tid; t < CHUNK * NOCT * 64; t += 128) ldst[t] = gsrc[t];
#endif
    };

    stage(0, 0);
    for (int ci = 0; ci < nch; ++ci) {
        const int which = ci & 1;
        if (ci + 1 < nch) {
            stage(ci + 1, which ^ 1);          // overlap next chunk with compute
#if SSD_HAVE_ASYNC_LDS
            __builtin_amdgcn_s_wait_asynccnt(2 * NOCT);  // current chunk landed
#endif
        } else {
#if SSD_HAVE_ASYNC_LDS
            __builtin_amdgcn_s_wait_asynccnt(0);
#endif
        }
        __syncthreads();
        __builtin_prefetch(inb + (size_t)((((ci + 1) * CHUNK) << 5) / 9) * HW, 0, 1);
#pragma unroll
        for (int kb2 = 0; kb2 < CHUNK; ++kb2) {
            const int kbase = (ci * CHUNK + kb2) << 5;
            // im2col gather: clamped unconditional loads + select (no EXEC churn)
            v16bf afrag;
#pragma unroll
            for (int j = 0; j < 16; ++j) {
                int kk = kbase + ((j >> 3) << 4) + hi8 + (j & 7);
                unsigned c   = (unsigned)kk / 9u;
                unsigned rem = (unsigned)kk - c * 9u;
                unsigned r3  = rem / 3u;
                unsigned s3  = rem - r3 * 3u;
                int yy = py + (int)r3 - 1;
                int xx = px + (int)s3 - 1;
                bool ok = pvalid && ((unsigned)yy < (unsigned)H) && ((unsigned)xx < (unsigned)W);
                int yyc = min(max(yy, 0), H - 1);
                int xxc = min(max(xx, 0), W - 1);
                float v = inb[c * HW + yyc * W + xxc];
                afrag[j] = f2bf(ok ? v : 0.0f);
            }
            // B fragments from LDS, software-pipelined one WMMA ahead so the
            // ds_load for o+1 overlaps the WMMA consuming fragment o.
            const unsigned short* wrow = &sbuf[which][kb2 * NOCT * 512 + lane * 16];
            v16bf bcur = *(const v16bf*)(wrow);
#pragma unroll
            for (int o = 0; o < NOCT; ++o) {
                v16bf bnext = (o + 1 < NOCT) ? *(const v16bf*)(wrow + (o + 1) * 512) : bcur;
                acc[o] = __builtin_amdgcn_wmma_f32_16x16x32_bf16(
                    false, afrag, false, bcur, (short)0, acc[o], false, false);
                bcur = bnext;
            }
        }
        __syncthreads();
    }

    // C/D layout: VGPR r -> M = r + 8*(lane/16), N = lane&15
#pragma unroll
    for (int o = 0; o < NOCT; ++o) {
        int oc = o * 16 + mrow;
        if (oc < OC) {
            float bval = bias[oc];
            int ka  = oc / ncols;
            int col = oc - ka * ncols;
#pragma unroll
            for (int r = 0; r < 8; ++r) {
                int pr = p0 + r + hi8;
                if (pr < P) {
                    int b2 = pr / HW;
                    int pix = pr - b2 * HW;
                    int anchor = anchor_base + pix * Kanch + ka;
                    out[((size_t)b2 * A + anchor) * ncols + col] = acc[o][r] + bval;
                }
            }
        }
    }
}

// ---- softmax + threshold + box decode -------------------------------------
__global__ void __launch_bounds__(256)
decode_k(const float* __restrict__ clsraw, const float* __restrict__ boxraw,
         const float* __restrict__ anc, float* __restrict__ scores,
         float* __restrict__ bboxes) {
    const int A = 8732, Bt = 32;
    int i = blockIdx.x * 256 + threadIdx.x;
    if (i >= Bt * A) return;
    int b = i / A, a = i - b * A;
    const float* cr = clsraw + (size_t)i * 21;
    float m = cr[0];
#pragma unroll
    for (int c = 1; c < 21; ++c) m = fmaxf(m, cr[c]);
    float e[21];
    float s = 0.0f;
#pragma unroll
    for (int c = 0; c < 21; ++c) { e[c] = __expf(cr[c] - m); s += e[c]; }
    float inv = 1.0f / s;
#pragma unroll
    for (int c = 0; c < 20; ++c) {
        float pr = e[c + 1] * inv;
        scores[((size_t)b * 20 + c) * A + a] = (pr > 0.01f) ? pr : 0.0f;
    }
    float acx = anc[a * 4 + 0], acy = anc[a * 4 + 1];
    float aw = anc[a * 4 + 2], ah = anc[a * 4 + 3];
    const float* br = boxraw + (size_t)i * 4;
    float cx = br[0] * 0.1f * aw + acx;
    float cy = br[1] * 0.1f * ah + acy;
    float w  = __expf(br[2] * 0.2f) * aw;
    float h  = __expf(br[3] * 0.2f) * ah;
    bboxes[(size_t)i * 4 + 0] = cx - 0.5f * w;
    bboxes[(size_t)i * 4 + 1] = cy - 0.5f * h;
    bboxes[(size_t)i * 4 + 2] = cx + 0.5f * w;
    bboxes[(size_t)i * 4 + 3] = cy + 0.5f * h;
}

// ---- per-image top-400 selection + class-aware serial NMS + emit ----------
__global__ void __launch_bounds__(256)
topk_nms_k(float* __restrict__ scores, const float* __restrict__ bboxes,
           float* __restrict__ outp) {
    const int A = 8732, NC = 20, NCAND = 400, NPOST = 100;
    const int b = blockIdx.x, tid = threadIdx.x;
    const int N = NC * A;
    float* sc = scores + (size_t)b * N;          // destroyed; rebuilt by decode_k
    const float* bx = bboxes + (size_t)b * A * 4;

    __shared__ float s_cls[NCAND], s_sc[NCAND];
    __shared__ float s_x1[NCAND], s_y1[NCAND], s_x2[NCAND], s_y2[NCAND];
    __shared__ int   s_keep[NCAND];
    __shared__ float red_v[256];
    __shared__ int   red_i[256];

    // top-400 by repeated argmax (stable ties -> lowest index, like lax.top_k)
    for (int t = 0; t < NCAND; ++t) {
        float bv = -1.0f;
        int bi = N;
        for (int i = tid; i < N; i += 256) {
            float v = sc[i];
            if (v > bv) { bv = v; bi = i; }
        }
        red_v[tid] = bv; red_i[tid] = bi;
        for (int off = 128; off > 0; off >>= 1) {
            __syncthreads();
            if (tid < off) {
                float v2 = red_v[tid + off]; int i2 = red_i[tid + off];
                if (v2 > red_v[tid] || (v2 == red_v[tid] && i2 < red_i[tid])) {
                    red_v[tid] = v2; red_i[tid] = i2;
                }
            }
        }
        __syncthreads();
        if (tid == 0) {
            int g = red_i[0];
            float v = red_v[0];
            int cls = g / A;
            int a   = g - cls * A;
            sc[g] = -1e30f;                       // exclude from later rounds
            bool valid = v > 0.0f;
            s_sc[t]  = valid ? v : 0.0f;
            s_cls[t] = valid ? (float)cls : -1.0f;
            s_x1[t] = bx[a * 4 + 0]; s_y1[t] = bx[a * 4 + 1];
            s_x2[t] = bx[a * 4 + 2]; s_y2[t] = bx[a * 4 + 3];
            s_keep[t] = valid ? 1 : 0;
        }
        __syncthreads();
    }

    // sequential class-aware NMS, IoU > 0.45 suppresses later candidates
    for (int i = 0; i < NCAND; ++i) {
        __syncthreads();
        if (!s_keep[i]) continue;                 // uniform branch
        float icls = s_cls[i];
        float ix1 = s_x1[i], iy1 = s_y1[i], ix2 = s_x2[i], iy2 = s_y2[i];
        float iarea = fmaxf(ix2 - ix1, 0.0f) * fmaxf(iy2 - iy1, 0.0f);
        for (int j = i + 1 + tid; j < NCAND; j += 256) {
            if (s_keep[j] && s_cls[j] == icls) {
                float w = fminf(ix2, s_x2[j]) - fmaxf(ix1, s_x1[j]);
                float h = fminf(iy2, s_y2[j]) - fmaxf(iy1, s_y1[j]);
                float inter = fmaxf(w, 0.0f) * fmaxf(h, 0.0f);
                float ja = fmaxf(s_x2[j] - s_x1[j], 0.0f) * fmaxf(s_y2[j] - s_y1[j], 0.0f);
                float iou = inter / fmaxf(iarea + ja - inter, 1e-12f);
                if (iou > 0.45f) s_keep[j] = 0;
            }
        }
    }
    __syncthreads();

    // emit first 100: tuple (ids[B,100,1], scores[B,100,1], boxes[B,100,4])
    float* oid = outp;
    float* osc = outp + 32 * 100;
    float* obx = outp + 2 * 32 * 100;
    for (int r = tid; r < NPOST; r += 256) {
        int k = s_keep[r];
        int o = b * 100 + r;
        oid[o] = k ? s_cls[r] : -1.0f;
        osc[o] = k ? s_sc[r] : -1.0f;
        obx[o * 4 + 0] = k ? s_x1[r] : -1.0f;
        obx[o * 4 + 1] = k ? s_y1[r] : -1.0f;
        obx[o * 4 + 2] = k ? s_x2[r] : -1.0f;
        obx[o * 4 + 3] = k ? s_y2[r] : -1.0f;
    }
}

// ---------------------------------------------------------------------------
extern "C" void kernel_launch(void* const* d_in, const int* in_sizes, int n_in,
                              void* d_out, int out_size, void* d_ws, size_t ws_size,
                              hipStream_t stream) {
    (void)in_sizes; (void)n_in; (void)out_size; (void)ws_size;
    static const int CH[6]   = {512, 1024, 512, 256, 256, 256};
    static const int FS[6]   = {38, 19, 10, 5, 3, 1};
    static const int KA[6]   = {4, 6, 6, 6, 4, 4};
    static const int BASE[6] = {0, 5776, 7942, 8542, 8692, 8728};
    const int A = 8732, Bt = 32;

    const float *feat[6], *cw[6], *cb[6], *bw[6], *bb[6];
    for (int i = 0; i < 6; i++) {
        feat[i] = (const float*)d_in[i];
        cw[i] = (const float*)d_in[6 + 4 * i + 0];
        cb[i] = (const float*)d_in[6 + 4 * i + 1];
        bw[i] = (const float*)d_in[6 + 4 * i + 2];
        bb[i] = (const float*)d_in[6 + 4 * i + 3];
    }

    char* wsb = (char*)d_ws;
    size_t cur = 0;
    auto carve = [&](size_t bytes) -> void* {
        void* p = wsb + cur;
        cur = (cur + bytes + 255) & ~(size_t)255;
        return p;
    };
    float* anc    = (float*)carve((size_t)A * 4 * sizeof(float));
    float* clsraw = (float*)carve((size_t)Bt * A * 21 * sizeof(float));
    float* boxraw = (float*)carve((size_t)Bt * A * 4 * sizeof(float));
    float* bboxes = (float*)carve((size_t)Bt * A * 4 * sizeof(float));
    float* scores = (float*)carve((size_t)Bt * 20 * A * sizeof(float));
    unsigned short* wpc[6];
    unsigned short* wpb[6];
    int noct_c[6], noct_b[6];
    for (int i = 0; i < 6; i++) {
        int nkb = CH[i] * 9 / 32;
        noct_c[i] = (KA[i] * 21 + 15) / 16;      // {6,8,8,8,6,6}
        noct_b[i] = (KA[i] * 4 + 15) / 16;       // {1,2,2,2,1,1}
        wpc[i] = (unsigned short*)carve((size_t)nkb * noct_c[i] * 512 * 2);
        wpb[i] = (unsigned short*)carve((size_t)nkb * noct_b[i] * 512 * 2);
    }

    // 1) pack weights into B-fragment order (bf16)
    for (int i = 0; i < 6; i++) {
        int nkb = CH[i] * 9 / 32;
        int tc = nkb * noct_c[i] * 512;
        int tb = nkb * noct_b[i] * 512;
        pack_weights_k<<<(tc + 255) / 256, 256, 0, stream>>>(cw[i], wpc[i], CH[i], KA[i] * 21, noct_c[i], tc);
        pack_weights_k<<<(tb + 255) / 256, 256, 0, stream>>>(bw[i], wpb[i], CH[i], KA[i] * 4, noct_b[i], tb);
    }

    // 2) anchors
    build_anchors_k<<<(A + 255) / 256, 256, 0, stream>>>(anc);

    // 3) WMMA convs, one pass per feature map per head
    for (int i = 0; i < 6; i++) {
        int P = Bt * FS[i] * FS[i];
        dim3 blk(32, 4, 1), grd((P + 63) / 64, 1, 1);
        switch (noct_c[i]) {
            case 6: conv3_wmma<6><<<grd, blk, 0, stream>>>(feat[i], wpc[i], cb[i], clsraw, CH[i], FS[i], FS[i], KA[i] * 21, KA[i], 21, BASE[i]); break;
            case 8: conv3_wmma<8><<<grd, blk, 0, stream>>>(feat[i], wpc[i], cb[i], clsraw, CH[i], FS[i], FS[i], KA[i] * 21, KA[i], 21, BASE[i]); break;
            default: break;
        }
        switch (noct_b[i]) {
            case 1: conv3_wmma<1><<<grd, blk, 0, stream>>>(feat[i], wpb[i], bb[i], boxraw, CH[i], FS[i], FS[i], KA[i] * 4, KA[i], 4, BASE[i]); break;
            case 2: conv3_wmma<2><<<grd, blk, 0, stream>>>(feat[i], wpb[i], bb[i], boxraw, CH[i], FS[i], FS[i], KA[i] * 4, KA[i], 4, BASE[i]); break;
            default: break;
        }
    }

    // 4) softmax + threshold + box decode
    int tot = Bt * A;
    decode_k<<<(tot + 255) / 256, 256, 0, stream>>>(clsraw, boxraw, anc, scores, bboxes);

    // 5) per-image top-400 + NMS + emit
    topk_nms_k<<<Bt, 256, 0, stream>>>(scores, bboxes, (float*)d_out);
}
